// CasualMultiHeadSelfAttention_63660005261369
// MI455X (gfx1250) — matrix operand order repaired and checked by reading, not compiled, not run on hardware
//
#include <hip/hip_runtime.h>
#include <hip/hip_bf16.h>

// ---------------------------------------------------------------------------
// Types for CDNA5 WMMA (wave32): v_wmma_f32_16x16x32_bf16
// ---------------------------------------------------------------------------
typedef __attribute__((ext_vector_type(16))) __bf16 v16bf;
typedef __attribute__((ext_vector_type(8)))  __bf16 v8bf;
typedef __attribute__((ext_vector_type(8)))  float  v8f;
typedef __attribute__((ext_vector_type(4)))  unsigned int v4u;
typedef __attribute__((ext_vector_type(8)))  int v8i;
typedef __attribute__((ext_vector_type(4)))  int v4i;

#define DMODEL 1024
#define NHEADS 16
#define DK     64
#define SEQ    2048
#define BATCH  2
#define NTOK   (BATCH * SEQ)        // 4096
#define LN_THETA 9.210340371976184f // ln(10000)

// ---------------------------------------------------------------------------
// Fragment loaders (layouts per CDNA5 ISA 7.12.2)
//
// A (16x32 bf16, MxK): lane l<16 -> row M=l, VGPR0-3: K=0..7, VGPR4-7: K=16..23
//                      lane l>=16 -> row M=l-16, K=8..15 and K=24..31
// B (32x16 bf16, KxN): lanes 0-15: K=0..7,16..23 / N=lane;
//                      lanes 16-31: K=8..15,24..31 / N=lane-16
// C/D (16x16 f32):     VGPR r, lanes 0-15: M=r, N=lane; lanes 16-31: M=r+8
// ---------------------------------------------------------------------------
__device__ __forceinline__ v16bf load_frag_a(const __bf16* base, int ld,
                                             int row0, int k0, int lane) {
  int r    = row0 + (lane & 15);
  int koff = k0 + ((lane & 16) ? 8 : 0);
  const __bf16* p = base + (size_t)r * ld + koff;
  v16bf a;
  *((v8bf*)&a)       = *(const v8bf*)(p);        // K 0..7   (or 8..15)
  *(((v8bf*)&a) + 1) = *(const v8bf*)(p + 16);   // K 16..23 (or 24..31)
  return a;
}

__device__ __forceinline__ v16bf load_frag_b(const __bf16* base, int ld,
                                             int col0, int k0, int lane) {
  int c    = col0 + (lane & 15);
  int koff = k0 + ((lane & 16) ? 8 : 0);
  const __bf16* p = base + (size_t)c * ld + koff;  // K 0..7,16..23 (or 8..15,24..31)
  v16bf b;
  *((v8bf*)&b)       = *(const v8bf*)(p);
  *(((v8bf*)&b) + 1) = *(const v8bf*)(p + 16);
  return b;
}

__device__ __forceinline__ v8f wmma_bf16(v16bf a, v16bf b, v8f c) {
  return __builtin_amdgcn_wmma_f32_16x16x32_bf16(false, a, false, b,
                                                 (short)0, c, false, false);
}

// ---------------------------------------------------------------------------
// TDM: async copy of a 64-row x 32-col bf16 tile of W (row stride DMODEL)
// into LDS, packed [64][32]. D# per CDNA5 ISA.
// This toolchain exposes the 6-arg builtin: (v4u, v8i, v4i, v4i, v8i, i32).
// ---------------------------------------------------------------------------
__device__ __forceinline__ void tdm_load_w(const __bf16* lds_dst,
                                           const __bf16* gsrc,
                                           int col0, int k0) {
  unsigned long long ga =
      (unsigned long long)(uintptr_t)(gsrc + (size_t)col0 * DMODEL + k0);
  unsigned int lds_off = (unsigned int)(uintptr_t)lds_dst;  // flat addr low 32 = LDS offset

  v4u g0;
  g0[0] = 1u;                                    // count=1, user descriptor
  g0[1] = lds_off;                               // lds_addr (bytes)
  g0[2] = (unsigned int)(ga & 0xffffffffu);      // global_addr[31:0]
  g0[3] = (unsigned int)((ga >> 32) & 0x01ffffffu) | (2u << 30); // addr[56:32] | type=2

  v8i g1;
  g1[0] = 0x00010000;                 // data_size=1 (2 bytes), no flags
  g1[1] = (int)((DMODEL & 0xffff) << 16);          // tensor_dim0 lo16
  g1[2] = (int)(((unsigned)DMODEL >> 16) | ((DMODEL & 0xffff) << 16)); // dim0 hi | dim1 lo
  g1[3] = (int)(((unsigned)DMODEL >> 16) | (32u << 16)); // dim1 hi | tile_dim0=32
  g1[4] = 64;                         // tile_dim1=64, tile_dim2=0
  g1[5] = DMODEL;                     // tensor_dim0_stride lo32
  g1[6] = 0;                          // stride0 hi | stride1 lo
  g1[7] = 0;                          // stride1 hi

  v4i g2 = {0, 0, 0, 0};
  v4i g3 = {0, 0, 0, 0};
  v8i g4 = {0, 0, 0, 0, 0, 0, 0, 0};
  __builtin_amdgcn_tensor_load_to_lds(g0, g1, g2, g3, g4, 0);
}

// ---------------------------------------------------------------------------
// fp32 -> bf16 cast
// ---------------------------------------------------------------------------
__global__ void cast_f32_bf16(const float* __restrict__ in,
                              __bf16* __restrict__ out, int n) {
  int i = blockIdx.x * blockDim.x + threadIdx.x;
  if (i < n) out[i] = (__bf16)in[i];
}

// ---------------------------------------------------------------------------
// GEMM: Y[n,e] = sum_d X[n,d] * W[e,d]   (X: NTOKxDMODEL, W: DMODELxDMODEL)
// mode 0: f32 store to outF
// mode 1: fused RoPE (per token position), bf16 store to outB
// mode 2: bf16 store transposed to outB as [b][h][dk][SEQ]  (for V)
// Wave tile: 32 rows x 64 cols; 8 waves stacked in M -> 256 rows/block.
// W tile (64x32) staged in LDS by TDM, double-buffered, s_wait_tensorcnt sync.
// ---------------------------------------------------------------------------
__global__ void __launch_bounds__(256)
gemm_bf16_kernel(const __bf16* __restrict__ X, const __bf16* __restrict__ W,
                 float* __restrict__ outF, __bf16* __restrict__ outB,
                 const int* __restrict__ pos, int mode) {
  __shared__ alignas(128) __bf16 wtile[2][64 * 32];  // 2 x 4KB

  const int lane = threadIdx.x & 31;
  const int wave = threadIdx.x >> 5;
  const int row0 = (blockIdx.x * 8 + wave) * 32;
  const int col0 = blockIdx.y * 64;

  v8f acc[2][4] = {};

  if (threadIdx.x < 32) tdm_load_w(&wtile[0][0], W, col0, 0);

  for (int k0 = 0, it = 0; k0 < DMODEL; k0 += 32, ++it) {
    const int cur = it & 1;
    if (threadIdx.x < 32) __builtin_amdgcn_s_wait_tensorcnt(0);
    __syncthreads();                      // wtile[cur] ready; wtile[cur^1] free
    if (threadIdx.x < 32 && (k0 + 32) < DMODEL)
      tdm_load_w(&wtile[cur ^ 1][0], W, col0, k0 + 32);

    if (k0 + 32 < DMODEL)
      __builtin_prefetch(X + (size_t)(row0 + (lane & 15)) * DMODEL + k0 + 32, 0, 3);

    v16bf a0 = load_frag_a(X, DMODEL, row0,      k0, lane);
    v16bf a1 = load_frag_a(X, DMODEL, row0 + 16, k0, lane);
#pragma unroll
    for (int t = 0; t < 4; ++t) {
      v16bf b = load_frag_b(&wtile[cur][0], 32, t * 16, 0, lane);
      acc[0][t] = wmma_bf16(a0, b, acc[0][t]);
      acc[1][t] = wmma_bf16(a1, b, acc[1][t]);
    }
  }

  const int lhalf = (lane & 16) ? 8 : 0;
  const int ln    = lane & 15;

#pragma unroll
  for (int s2 = 0; s2 < 2; ++s2) {
#pragma unroll
    for (int t = 0; t < 4; ++t) {
#pragma unroll
      for (int r = 0; r < 8; ++r) {
        const int m   = row0 + s2 * 16 + lhalf + r;
        const int col = col0 + t * 16 + ln;
        float v = acc[s2][t][r];
        if (mode == 0) {
          outF[(size_t)m * DMODEL + col] = v;
        } else if (mode == 1) {
          // RoPE: even/odd columns are adjacent lanes in C layout.
          float partner = __shfl_xor(v, 1, 32);
          float p    = (float)pos[m];
          float freq = __expf(-(float)(col & 62) * (LN_THETA / (float)DK));
          float ang  = p * freq;
          float cs   = __cosf(ang);
          float sn   = __sinf(ang);
          float res  = (col & 1) ? (partner * sn + v * cs)   // odd:  xe*sin + xo*cos
                                 : (v * cs - partner * sn);  // even: xe*cos - xo*sin
          outB[(size_t)m * DMODEL + col] = (__bf16)res;
        } else {
          // V transposed: [b][h][dk][SEQ]
          int h  = col >> 6, dl = col & 63;
          int bb = m >> 11,  s  = m & (SEQ - 1);
          outB[(((size_t)(bb * NHEADS + h) * DK + dl) * SEQ) + s] = (__bf16)v;
        }
      }
    }
  }
}

// ---------------------------------------------------------------------------
// Causal flash attention. One wave = one (b, h, 16-query tile).
// Scores via WMMA, online softmax with lane-half shuffle reductions,
// P staged through per-wave LDS to reach the A layout, P.V via WMMA vs V^T.
// ---------------------------------------------------------------------------
__global__ void __launch_bounds__(128)
attn_kernel(const __bf16* __restrict__ Qb, const __bf16* __restrict__ Kb,
            const __bf16* __restrict__ Vt, __bf16* __restrict__ ctx) {
  __shared__ alignas(64) __bf16 plds[4][16 * 32];

  const int lane = threadIdx.x & 31;
  const int wave = threadIdx.x >> 5;
  const int flat = blockIdx.x * 4 + wave;
  const int qt = flat & 127;             // 128 query tiles
  const int h  = (flat >> 7) & (NHEADS - 1);
  const int b  = flat >> 11;
  const int q0 = qt * 16;

  const __bf16* Qbase = Qb + (size_t)b * SEQ * DMODEL + h * DK;
  const __bf16* Kbase = Kb + (size_t)b * SEQ * DMODEL + h * DK;
  const __bf16* Vbase = Vt + (size_t)(b * NHEADS + h) * DK * SEQ;
  __bf16* pbuf = plds[wave];

  v16bf aq0 = load_frag_a(Qbase, DMODEL, q0, 0,  lane);
  v16bf aq1 = load_frag_a(Qbase, DMODEL, q0, 32, lane);

  v8f acc[4] = {};
  float mrow[8], lrow[8];
#pragma unroll
  for (int r = 0; r < 8; ++r) { mrow[r] = -1e30f; lrow[r] = 0.0f; }

  const int lhalf = (lane & 16) ? 8 : 0;
  const int ln    = lane & 15;

  for (int k0 = 0; k0 < q0 + 16; k0 += 32) {
    // ---- scores: two 16x16 key tiles, contraction over dk in 2 steps ----
    v8f sc[2];
#pragma unroll
    for (int t = 0; t < 2; ++t) {
      v16bf bk0 = load_frag_b(Kbase, DMODEL, k0 + t * 16, 0,  lane);
      v16bf bk1 = load_frag_b(Kbase, DMODEL, k0 + t * 16, 32, lane);
      v8f z = {};
      z = wmma_bf16(aq0, bk0, z);
      z = wmma_bf16(aq1, bk1, z);
      sc[t] = z;
    }

    // ---- online softmax over the 32 keys of this step ----
    const int key0 = k0 + ln;
    const int key1 = k0 + 16 + ln;
#pragma unroll
    for (int r = 0; r < 8; ++r) {
      const int row = q0 + lhalf + r;
      float s0 = sc[0][r] * 0.125f;           // 1/sqrt(64)
      float s1 = sc[1][r] * 0.125f;
      if (key0 > row) s0 = -1e30f;            // causal mask
      if (key1 > row) s1 = -1e30f;

      float mx = fmaxf(s0, s1);
#pragma unroll
      for (int off = 1; off < 16; off <<= 1)
        mx = fmaxf(mx, __shfl_xor(mx, off, 32));

      float mnew  = fmaxf(mrow[r], mx);
      float alpha = __expf(mrow[r] - mnew);
      float p0 = __expf(s0 - mnew);
      float p1 = __expf(s1 - mnew);
      float sum = p0 + p1;
#pragma unroll
      for (int off = 1; off < 16; off <<= 1)
        sum += __shfl_xor(sum, off, 32);

      lrow[r] = lrow[r] * alpha + sum;
      mrow[r] = mnew;
#pragma unroll
      for (int t = 0; t < 4; ++t) acc[t][r] *= alpha;

      // stage P (C layout -> LDS row-major [16 rows][32 keys])
      pbuf[(lhalf + r) * 32 + ln]      = (__bf16)p0;
      pbuf[(lhalf + r) * 32 + 16 + ln] = (__bf16)p1;
    }

    asm volatile("s_wait_dscnt 0" ::: "memory");   // P stores visible to our DS loads

    // ---- P (A layout from LDS) x V^T ----
    v16bf pa = load_frag_a(pbuf, 32, 0, 0, lane);
#pragma unroll
    for (int t = 0; t < 4; ++t) {
      v16bf bv = load_frag_b(Vbase, SEQ, t * 16, k0, lane);
      acc[t] = wmma_bf16(pa, bv, acc[t]);
    }
    asm volatile("s_wait_dscnt 0" ::: "memory");   // P loads done before next overwrite
  }

  // ---- normalize, store context [token][DMODEL] in bf16 ----
#pragma unroll
  for (int t = 0; t < 4; ++t) {
#pragma unroll
    for (int r = 0; r < 8; ++r) {
      const int row = q0 + lhalf + r;
      const int n   = b * SEQ + row;
      const int col = h * DK + t * 16 + ln;
      float v = acc[t][r] / lrow[r];
      ctx[(size_t)n * DMODEL + col] = (__bf16)v;
    }
  }
}

// ---------------------------------------------------------------------------
// Host-side orchestration
// ---------------------------------------------------------------------------
extern "C" void kernel_launch(void* const* d_in, const int* in_sizes, int n_in,
                              void* d_out, int out_size, void* d_ws, size_t ws_size,
                              hipStream_t stream) {
  (void)in_sizes; (void)n_in; (void)out_size; (void)ws_size;

  const float* x  = (const float*)d_in[0];
  const float* Wq = (const float*)d_in[1];
  const float* Wk = (const float*)d_in[2];
  const float* Wv = (const float*)d_in[3];
  const float* Wo = (const float*)d_in[4];
  const int*  pos = (const int*)d_in[5];
  float* out = (float*)d_out;

  char* w = (char*)d_ws;
  const size_t XB_BYTES = (size_t)NTOK * DMODEL * 2;     // 8 MB
  const size_t WB_BYTES = (size_t)DMODEL * DMODEL * 2;   // 2 MB
  __bf16* xb  = (__bf16*)(w);
  __bf16* Wqb = (__bf16*)(w + XB_BYTES);
  __bf16* Wkb = (__bf16*)(w + XB_BYTES + 1 * WB_BYTES);
  __bf16* Wvb = (__bf16*)(w + XB_BYTES + 2 * WB_BYTES);
  __bf16* Wob = (__bf16*)(w + XB_BYTES + 3 * WB_BYTES);
  __bf16* Qb  = (__bf16*)(w + XB_BYTES + 4 * WB_BYTES);
  __bf16* Kb  = (__bf16*)(w + 2 * XB_BYTES + 4 * WB_BYTES);
  __bf16* Vt  = (__bf16*)(w + 3 * XB_BYTES + 4 * WB_BYTES);
  __bf16* ctx = (__bf16*)(w + 4 * XB_BYTES + 4 * WB_BYTES);

  const int nX = NTOK * DMODEL;      // 4194304
  const int nW = DMODEL * DMODEL;    // 1048576
  cast_f32_bf16<<<(nX + 255) / 256, 256, 0, stream>>>(x,  xb,  nX);
  cast_f32_bf16<<<(nW + 255) / 256, 256, 0, stream>>>(Wq, Wqb, nW);
  cast_f32_bf16<<<(nW + 255) / 256, 256, 0, stream>>>(Wk, Wkb, nW);
  cast_f32_bf16<<<(nW + 255) / 256, 256, 0, stream>>>(Wv, Wvb, nW);
  cast_f32_bf16<<<(nW + 255) / 256, 256, 0, stream>>>(Wo, Wob, nW);

  dim3 gg(NTOK / 256, DMODEL / 64);  // (16, 16)
  gemm_bf16_kernel<<<gg, 256, 0, stream>>>(xb, Wqb, nullptr, Qb, pos, 1); // Q + RoPE
  gemm_bf16_kernel<<<gg, 256, 0, stream>>>(xb, Wkb, nullptr, Kb, pos, 1); // K + RoPE
  gemm_bf16_kernel<<<gg, 256, 0, stream>>>(xb, Wvb, nullptr, Vt, nullptr, 2); // V^T

  attn_kernel<<<(BATCH * NHEADS * (SEQ / 16)) / 4, 128, 0, stream>>>(Qb, Kb, Vt, ctx);

  gemm_bf16_kernel<<<gg, 256, 0, stream>>>(ctx, Wob, out, nullptr, nullptr, 0); // Wo proj
}
